// PoiEnhancer_25537875542494
// MI455X (gfx1250) — compile-verified, matrix operand order checked by reading, not run on hardware
//
#include <hip/hip_runtime.h>
#include <hip/hip_fp16.h>

typedef __attribute__((ext_vector_type(2)))  float    v2f;
typedef __attribute__((ext_vector_type(4)))  float    v4f;
typedef __attribute__((ext_vector_type(8)))  float    v8f;
typedef __attribute__((ext_vector_type(2)))  _Float16 v2h;
typedef __attribute__((ext_vector_type(16))) _Float16 v16h;

#define ROWS 8192            // B*N
#define DIM  256             // POI_DIM
#define SZ   (ROWS * DIM)    // 2,097,152 floats

// ---------------------------------------------------------------------------
// One-time f32 -> f16 conversion of the hidden-projection weights (2 tables)
// ---------------------------------------------------------------------------
__global__ __launch_bounds__(256)
void convert_w_kernel(const float* __restrict__ Wf, _Float16* __restrict__ Wh)
{
    const int i = blockIdx.x * 256 + threadIdx.x;   // grid = 2*256*4096/256
    Wh[i] = (_Float16)Wf[i];
}

// ---------------------------------------------------------------------------
// Embedding GEMM: H[row,:] = f16(table[batch[row]]) (1x4096) @ Wh^T (256x4096)
// f16 WMMA 16x16x32, one 16x16 C tile per wave, 8 waves -> 32x64 block tile.
// A rows are gathered on the fly (never materialized); weights are the
// pre-converted f16 copy.
// ---------------------------------------------------------------------------
__global__ __launch_bounds__(256)
void embed_gemm_kernel(const int* __restrict__ batch,
                       const _Float16* __restrict__ table,   // NUM x 4096 (f16)
                       const _Float16* __restrict__ Wh,      // 256 x 4096 (f16)
                       float* __restrict__ H)                // 8192 x 256
{
    const int wave = threadIdx.x >> 5;
    const int lane = threadIdx.x & 31;
    const int half = lane >> 4;
    const int r    = lane & 15;
    const int tileN = (blockIdx.x * 4 + (wave & 3)) * 16;
    const int tileM = (blockIdx.y * 2 + (wave >> 2)) * 16;

    const _Float16* arow = table + (size_t)batch[tileM + r] * 4096;
    const _Float16* wrow = Wh + (size_t)(tileN + r) * 4096;

    v8f acc = {};
    for (int k = 0; k < 4096; k += 32) {
        if (k + 512 < 4096)                           // stream-ahead prefetch
            __builtin_prefetch(arow + k + 512, 0, 3); // -> global_prefetch_b8
        v16h a, b;
#pragma unroll
        for (int v = 0; v < 8; ++v) {
            // A/B 16x32 f16 layout: VGPR v holds K pair {K0,K0+1},
            // K0 = 2v (+8 for v>=4) + 8*half
            const int k0 = k + 2 * v + ((v >= 4) ? 8 : 0) + 8 * half;
            v2h ap = *(const v2h*)(arow + k0);
            a[2 * v]     = ap.x;
            a[2 * v + 1] = ap.y;
            v2h bp = *(const v2h*)(wrow + k0);
            b[2 * v]     = bp.x;
            b[2 * v + 1] = bp.y;
        }
        acc = __builtin_amdgcn_wmma_f32_16x16x32_f16(false, a, false, b,
                                                     (short)0, acc, false, false);
    }
#pragma unroll
    for (int v = 0; v < 8; ++v)
        H[(size_t)(tileM + v + 8 * half) * DIM + tileN + r] = acc[v];
}

// ---------------------------------------------------------------------------
// L2 row-normalize (wave per row of 256)
// ---------------------------------------------------------------------------
__global__ __launch_bounds__(256)
void rownorm_kernel(float* __restrict__ H)
{
    const int wave = threadIdx.x >> 5, lane = threadIdx.x & 31;
    const int row  = blockIdx.x * 8 + wave;
    float* x = H + (size_t)row * DIM;
    float v[8], s = 0.f;
#pragma unroll
    for (int i = 0; i < 8; ++i) { v[i] = x[lane + 32 * i]; s += v[i] * v[i]; }
#pragma unroll
    for (int off = 16; off > 0; off >>= 1) s += __shfl_xor(s, off, 32);
    const float inv = 1.0f / fmaxf(sqrtf(s), 1e-12f);
#pragma unroll
    for (int i = 0; i < 8; ++i) x[lane + 32 * i] = v[i] * inv;
}

// ---------------------------------------------------------------------------
// LayerNorm with gamma (wave per row of 256)
// ---------------------------------------------------------------------------
__global__ __launch_bounds__(256)
void ln_kernel(const float* __restrict__ X, const float* __restrict__ gamma,
               float* __restrict__ O)
{
    const int wave = threadIdx.x >> 5, lane = threadIdx.x & 31;
    const int row  = blockIdx.x * 8 + wave;
    const float* x = X + (size_t)row * DIM;
    float v[8], s = 0.f;
#pragma unroll
    for (int i = 0; i < 8; ++i) { v[i] = x[lane + 32 * i]; s += v[i]; }
#pragma unroll
    for (int off = 16; off > 0; off >>= 1) s += __shfl_xor(s, off, 32);
    const float mu = s * (1.0f / 256.0f);
    float var = 0.f;
#pragma unroll
    for (int i = 0; i < 8; ++i) { const float d = v[i] - mu; var += d * d; }
#pragma unroll
    for (int off = 16; off > 0; off >>= 1) var += __shfl_xor(var, off, 32);
    const float inv = rsqrtf(var * (1.0f / 256.0f) + 1e-5f);
#pragma unroll
    for (int i = 0; i < 8; ++i)
        O[(size_t)row * DIM + lane + 32 * i] =
            gamma[lane + 32 * i] * (v[i] - mu) * inv;
}

// ---------------------------------------------------------------------------
// Generic f32 WMMA GEMM: C(MxN,ldc) = alpha * A(MxK) @ W(NxK)^T   [+= if ACCUM]
// grid: (ceil(N/64), M/32), block 256 (8 waves, each a 16x16 tile)
// ---------------------------------------------------------------------------
template <bool ACCUM>
__global__ __launch_bounds__(256)
void gemm_f32_kernel(const float* __restrict__ A, const float* __restrict__ W,
                     float* __restrict__ C, int N, int K, int ldc, float alpha)
{
    const int wave = threadIdx.x >> 5, lane = threadIdx.x & 31;
    const int half = lane >> 4, r = lane & 15;
    const int tileN = (blockIdx.x * 4 + (wave & 3)) * 16;
    const int tileM = (blockIdx.y * 2 + (wave >> 2)) * 16;
    if (tileN >= N) return;                 // wave-uniform

    const float* arow = A + (size_t)(tileM + r) * K;
    const float* wrow = W + (size_t)(tileN + r) * K;
    v8f acc = {};
    for (int k = 0; k < K; k += 4) {
        v2f a = *(const v2f*)(arow + k + 2 * half);
        v2f b = *(const v2f*)(wrow + k + 2 * half);
        acc = __builtin_amdgcn_wmma_f32_16x16x4_f32(false, a, false, b,
                                                    (short)0, acc, false, false);
    }
#pragma unroll
    for (int v = 0; v < 8; ++v) {
        const size_t idx = (size_t)(tileM + v + 8 * half) * ldc + tileN + r;
        const float val = acc[v] * alpha;
        C[idx] = ACCUM ? (C[idx] + val) : val;
    }
}

// ---------------------------------------------------------------------------
// Fused SwiGLU up-projection: FFP = silu(XN@W1[1024:].T) * (XN@W1[:1024].T)
// ---------------------------------------------------------------------------
__global__ __launch_bounds__(256)
void ffn_gemm_kernel(const float* __restrict__ XN, const float* __restrict__ W1,
                     float* __restrict__ FFP)       // 8192 x 1024
{
    const int wave = threadIdx.x >> 5, lane = threadIdx.x & 31;
    const int half = lane >> 4, r = lane & 15;
    const int tileN = (blockIdx.x * 4 + (wave & 3)) * 16;   // grid.x = 16
    const int tileM = (blockIdx.y * 2 + (wave >> 2)) * 16;

    const float* arow = XN + (size_t)(tileM + r) * DIM;
    const float* w1h  = W1 + (size_t)(tileN + r) * DIM;          // h1 rows
    const float* w1g  = W1 + (size_t)(1024 + tileN + r) * DIM;   // gate rows
    v8f acc1 = {}, accg = {};
    for (int k = 0; k < DIM; k += 4) {
        v2f a  = *(const v2f*)(arow + k + 2 * half);
        v2f b1 = *(const v2f*)(w1h + k + 2 * half);
        v2f bg = *(const v2f*)(w1g + k + 2 * half);
        acc1 = __builtin_amdgcn_wmma_f32_16x16x4_f32(false, a, false, b1,
                                                     (short)0, acc1, false, false);
        accg = __builtin_amdgcn_wmma_f32_16x16x4_f32(false, a, false, bg,
                                                     (short)0, accg, false, false);
    }
#pragma unroll
    for (int v = 0; v < 8; ++v) {
        const float g = accg[v];
        const float s = g / (1.0f + __expf(-g));           // silu(gate)
        FFP[(size_t)(tileM + v + 8 * half) * 1024 + tileN + r] = s * acc1[v];
    }
}

// ---------------------------------------------------------------------------
// Attention: per (b, head). V tile is staged to LDS with async-to-LDS copies
// (ASYNCcnt) overlapped with the sim WMMAs; sim = Qh(128x32)@K^T (WMMA) ->
// LDS, row softmax, out = attn(128x128)@V (WMMA, V from LDS).
// Q pre-scaled by dim_head^-0.5.
// ---------------------------------------------------------------------------
__global__ __launch_bounds__(256)
void attention_kernel(const float* __restrict__ Q, const float* __restrict__ KV,
                      float* __restrict__ AO, int heads)
{
    __shared__ float sS[128 * 128];                // 64 KB score tile
    __shared__ float sV[128 * 32];                 // 16 KB V tile
    const int b = blockIdx.x / heads;
    const int h = blockIdx.x % heads;
    const int wave = threadIdx.x >> 5, lane = threadIdx.x & 31;
    const int half = lane >> 4, r = lane & 15;
    const int nq = heads * 32;

    // ---- async V staging: 1024 x 16B chunks, 4 per thread ----------------
#pragma unroll
    for (int i = 0; i < 4; ++i) {
        const int c    = threadIdx.x * 4 + i;      // chunk id 0..1023
        const int row  = c >> 3;                   // 8 chunks per 32-f row
        const int col4 = (c & 7) * 4;
        const unsigned long long ga =
            (unsigned long long)(const void*)(KV + ((size_t)b * 128 + row) * 64 + 32 + col4);
        const unsigned lds = (unsigned)(size_t)(const void*)(sV + row * 32 + col4);
        asm volatile("global_load_async_to_lds_b128 %0, %1, off"
                     :: "v"(lds), "v"(ga) : "memory");
    }

    // ---- sim = Q K^T (overlaps with async copies) ------------------------
    const float* qrow = Q + ((size_t)b * 128 + wave * 16 + r) * nq + h * 32;
    for (int tn = 0; tn < 8; ++tn) {
        const float* krow = KV + ((size_t)b * 128 + tn * 16 + r) * 64;
        v8f acc = {};
#pragma unroll
        for (int k = 0; k < 32; k += 4) {
            v2f a  = *(const v2f*)(qrow + k + 2 * half);
            v2f bb = *(const v2f*)(krow + k + 2 * half);
            acc = __builtin_amdgcn_wmma_f32_16x16x4_f32(false, a, false, bb,
                                                        (short)0, acc, false, false);
        }
#pragma unroll
        for (int v = 0; v < 8; ++v)
            sS[(wave * 16 + v + 8 * half) * 128 + tn * 16 + r] = acc[v];
    }
    asm volatile("s_wait_asynccnt 0x0" ::: "memory");  // V tile landed
    __syncthreads();

    if (threadIdx.x < 128) {                       // amax-shifted softmax row
        v4f* rowp = (v4f*)(sS + threadIdx.x * 128);
        v4f m4 = rowp[0];
        for (int j = 1; j < 32; ++j) {
            const v4f t = rowp[j];
            m4.x = fmaxf(m4.x, t.x); m4.y = fmaxf(m4.y, t.y);
            m4.z = fmaxf(m4.z, t.z); m4.w = fmaxf(m4.w, t.w);
        }
        const float m = fmaxf(fmaxf(m4.x, m4.y), fmaxf(m4.z, m4.w));
        float ssum = 0.f;
        for (int j = 0; j < 32; ++j) {
            v4f t = rowp[j];
            t.x = __expf(t.x - m); t.y = __expf(t.y - m);
            t.z = __expf(t.z - m); t.w = __expf(t.w - m);
            ssum += t.x + t.y + t.z + t.w;
            rowp[j] = t;
        }
        const float inv = 1.0f / ssum;
        for (int j = 0; j < 32; ++j) {
            v4f t = rowp[j];
            t.x *= inv; t.y *= inv; t.z *= inv; t.w *= inv;
            rowp[j] = t;
        }
    }
    __syncthreads();

    // ---- out = attn @ V (V from LDS) -------------------------------------
    const float* srow = sS + (wave * 16 + r) * 128;
    for (int tn = 0; tn < 2; ++tn) {
        v8f acc = {};
        for (int k = 0; k < 128; k += 4) {
            const int kb = k + 2 * half;
            v2f a = *(const v2f*)(srow + kb);
            v2f bb;
            bb.x = sV[kb * 32 + tn * 16 + r];
            bb.y = sV[(kb + 1) * 32 + tn * 16 + r];
            acc = __builtin_amdgcn_wmma_f32_16x16x4_f32(false, a, false, bb,
                                                        (short)0, acc, false, false);
        }
#pragma unroll
        for (int v = 0; v < 8; ++v)
            AO[((size_t)b * 128 + wave * 16 + v + 8 * half) * nq + h * 32 + tn * 16 + r] = acc[v];
    }
}

// ---------------------------------------------------------------------------
// y gather (fp16 -> f32)
// ---------------------------------------------------------------------------
__global__ __launch_bounds__(256)
void gather_y_kernel(const int* __restrict__ batch,
                     const _Float16* __restrict__ poi, float* __restrict__ Y)
{
    const int i = blockIdx.x * 256 + threadIdx.x;     // grid = SZ/256
    const int row = i >> 8, c = i & 255;
    Y[i] = (float)poi[(size_t)batch[row] * DIM + c];
}

// ---------------------------------------------------------------------------
// query / positive / negative views of z
// ---------------------------------------------------------------------------
__global__ __launch_bounds__(256)
void tails_kernel(const float* __restrict__ z, float* __restrict__ q,
                  float* __restrict__ p, float* __restrict__ neg)
{
    const int i = blockIdx.x * 256 + threadIdx.x;     // grid covers 64*127*1536
    if (i < 64 * 1536) {
        const int b = i / 1536, c = i % 1536;
        q[i] = z[((size_t)b * 128 + 0) * 1536 + c];
        p[i] = z[((size_t)b * 128 + 1) * 1536 + c];
    }
    const int c = i % 1536, t = i / 1536;
    const int n = t % 127 + 1, b = t / 127;
    neg[i] = z[((size_t)b * 128 + n) * 1536 + c];
}

// ---------------------------------------------------------------------------
// Host-side: one attention block (shared math of self / cross / fused)
// ---------------------------------------------------------------------------
static void run_block(hipStream_t s,
                      const float* x, const float* y,
                      const float* gx, const float* gy,
                      const float* wq, const float* wkv, const float* wout,
                      const float* w1, const float* w2,
                      int heads, float* dest, int ldc,
                      float* XN, float* YN, float* Qb, float* KVb,
                      float* AOb, float* FFPb, bool yn_ready)
{
    const float scale = 0.17677669529663687f;   // 32^-0.5
    ln_kernel<<<ROWS / 8, 256, 0, s>>>(x, gx, XN);
    const float* yn = XN;
    if (y != x || gy != gx) {
        if (!yn_ready) ln_kernel<<<ROWS / 8, 256, 0, s>>>(y, gy, YN);
        yn = YN;
    }
    const int NQ = heads * 32;
    gemm_f32_kernel<false><<<dim3((NQ + 63) / 64, ROWS / 32), 256, 0, s>>>(
        XN, wq, Qb, NQ, DIM, NQ, scale);
    gemm_f32_kernel<false><<<dim3(1, ROWS / 32), 256, 0, s>>>(
        yn, wkv, KVb, 64, DIM, 64, 1.0f);
    attention_kernel<<<64 * heads, 256, 0, s>>>(Qb, KVb, AOb, heads);
    gemm_f32_kernel<false><<<dim3(4, ROWS / 32), 256, 0, s>>>(
        AOb, wout, dest, DIM, NQ, ldc, 1.0f);
    ffn_gemm_kernel<<<dim3(16, ROWS / 32), 256, 0, s>>>(XN, w1, FFPb);
    gemm_f32_kernel<true><<<dim3(4, ROWS / 32), 256, 0, s>>>(
        FFPb, w2, dest, DIM, 1024, ldc, 1.0f);      // dest = attn_out + ff
}

extern "C" void kernel_launch(void* const* d_in, const int* in_sizes, int n_in,
                              void* d_out, int out_size, void* d_ws, size_t ws_size,
                              hipStream_t stream)
{
    (void)in_sizes; (void)n_in; (void)out_size; (void)ws_size;
    const int*      batch   = (const int*)d_in[0];
    const _Float16* llm     = (const _Float16*)d_in[1];
    const _Float16* poi     = (const _Float16*)d_in[2];
    const float* w_hidden = (const float*)d_in[3];
    const float* sa_gamma = (const float*)d_in[4];
    const float* sa_wq    = (const float*)d_in[5];
    const float* sa_wkv   = (const float*)d_in[6];
    const float* sa_wout  = (const float*)d_in[7];
    const float* sa_w1    = (const float*)d_in[8];
    const float* sa_w2    = (const float*)d_in[9];
    const float* ca_gx    = (const float*)d_in[10];
    const float* ca_gy    = (const float*)d_in[11];
    const float* ca_wq    = (const float*)d_in[12];
    const float* ca_wkv   = (const float*)d_in[13];
    const float* ca_wout  = (const float*)d_in[14];
    const float* ca_w1    = (const float*)d_in[15];
    const float* ca_w2    = (const float*)d_in[16];
    const float* fa_gx    = (const float*)d_in[17];
    const float* fa_gy    = (const float*)d_in[18];
    const float* fa_wq    = (const float*)d_in[19];
    const float* fa_wkv   = (const float*)d_in[20];
    const float* fa_wout  = (const float*)d_in[21];
    const float* fa_w1    = (const float*)d_in[22];
    const float* fa_w2    = (const float*)d_in[23];

    float* ws  = (float*)d_ws;
    float* H0  = ws;
    float* H1  = ws + 1 * (size_t)SZ;
    float* X1  = ws + 2 * (size_t)SZ;
    float* X2  = ws + 3 * (size_t)SZ;
    float* Y   = ws + 4 * (size_t)SZ;
    float* XC  = ws + 5 * (size_t)SZ;
    float* XN  = ws + 6 * (size_t)SZ;
    float* YN  = ws + 7 * (size_t)SZ;
    float* Qb  = ws + 8 * (size_t)SZ;
    float* AOb = ws + 9 * (size_t)SZ;
    float* KVb = ws + 10 * (size_t)SZ;     // uses SZ/4
    float* FFP = ws + 11 * (size_t)SZ;     // uses 4*SZ
    _Float16* WhH = (_Float16*)(ws + 15 * (size_t)SZ);   // 2 x 256 x 4096 f16

    float* z    = (float*)d_out;           // (64,128,1536)
    float* outq = z + (size_t)64 * 128 * 1536;
    float* outp = outq + (size_t)64 * 1536;
    float* outn = outp + (size_t)64 * 1536;

    // 0) convert hidden-proj weights (tables 0,1) to f16 once
    convert_w_kernel<<<(2 * 256 * 4096) / 256, 256, 0, stream>>>(w_hidden, WhH);

    // 1) embeddings for tables 0,1 (table 2 never reaches the output)
    for (int t = 0; t < 2; ++t) {
        float* H = t ? H1 : H0;
        embed_gemm_kernel<<<dim3(4, ROWS / 32), 256, 0, stream>>>(
            batch, llm + (size_t)t * 20000 * 4096,
            WhH + (size_t)t * 256 * 4096, H);
        rownorm_kernel<<<ROWS / 8, 256, 0, stream>>>(H);
    }
    gather_y_kernel<<<SZ / 256, 256, 0, stream>>>(batch, poi, Y);

    // 2) self-attention blocks 0,1
    run_block(stream, H0, H0, sa_gamma, sa_gamma,
              sa_wq, sa_wkv, sa_wout, sa_w1, sa_w2,
              8, X1, DIM, XN, YN, Qb, KVb, AOb, FFP, false);
    run_block(stream, H1, H1, sa_gamma + 256, sa_gamma + 256,
              sa_wq + (size_t)1 * 256 * 256, sa_wkv + (size_t)1 * 64 * 256,
              sa_wout + (size_t)1 * 256 * 256, sa_w1 + (size_t)1 * 2048 * 256,
              sa_w2 + (size_t)1 * 256 * 1024,
              8, X2, DIM, XN, YN, Qb, KVb, AOb, FFP, false);

    // 3) cross block j -> XC, then fused block j -> z[:,:,j*256:(j+1)*256]
    for (int j = 0; j < 6; ++j) {
        run_block(stream, X1, X2, ca_gx + j * 256, ca_gy + j * 256,
                  ca_wq + (size_t)j * 256 * 256, ca_wkv + (size_t)j * 64 * 256,
                  ca_wout + (size_t)j * 256 * 256, ca_w1 + (size_t)j * 2048 * 256,
                  ca_w2 + (size_t)j * 256 * 1024,
                  8, XC, DIM, XN, YN, Qb, KVb, AOb, FFP, false);
        ln_kernel<<<ROWS / 8, 256, 0, stream>>>(Y, fa_gy, YN);
        run_block(stream, XC, Y, fa_gx, fa_gy,
                  fa_wq, fa_wkv, fa_wout, fa_w1, fa_w2,
                  1, z + (size_t)j * 256, 6 * DIM,
                  XN, YN, Qb, KVb, AOb, FFP, true);
    }

    // 4) query / positive / negative
    tails_kernel<<<(64 * 127 * 1536) / 256, 256, 0, stream>>>(z, outq, outp, outn);
}
